// Conv2DQwenRMSNorm_36318243455326
// MI455X (gfx1250) — matrix operand order repaired and checked by reading, not compiled, not run on hardware
//
#include <hip/hip_runtime.h>
#include <hip/hip_bf16.h>

// Problem constants from the reference: B=4, S=4096, H=1024, L=20.
#define HDIM    1024
#define NLAYER  20
#define ROWS    16384                  // B*S output rows
#define EPS     1e-06f
#define KTILES  (HDIM / 32)            // 32 K-steps of 32
#define NSUB    (HDIM / 16)            // 64 N-subtiles of 16

typedef __attribute__((ext_vector_type(16))) __bf16    v16bf;
typedef __attribute__((ext_vector_type(8)))  float     v8f;
typedef __attribute__((ext_vector_type(8)))  unsigned  v8u;

// round-to-nearest-even f32 -> bf16, packed pair {lo = a, hi = b}
__device__ __forceinline__ unsigned pack_bf16(float a, float b) {
    union { float f; unsigned u; } ua, ub;
    ua.f = a; ub.f = b;
    unsigned ha = (ua.u + 0x7FFFu + ((ua.u >> 16) & 1u)) >> 16;
    unsigned hb = (ub.u + 0x7FFFu + ((ub.u >> 16) & 1u)) >> 16;
    return (ha & 0xFFFFu) | (hb << 16);
}

// ---------------------------------------------------------------------------
// Kernel 1: build the WMMA B operand in *fragment-major* layout.
// wfrag[(((ktile*NSUB + subN)*32 + lane)*8) + v] holds the packed bf16 pair
// a lane needs for B-VGPR v of subtile (ktile, subN):
//   col o = subN*16 + (lane&15),  kpair = ktile*16 + (v>>2)*8 + (lane>>4)*4 + (v&3)
//   value = pack( (1/L)*sum_l W[l][o][2kp], (1/L)*sum_l W[l][o][2kp+1] )
// GEMM then fetches each 8-dword fragment slice with two global_load_b128.
// ---------------------------------------------------------------------------
__global__ void reduce_w_kernel(const float* __restrict__ conv_w,
                                unsigned* __restrict__ wfrag) {
    const int tid  = blockIdx.x * blockDim.x + threadIdx.x;  // 0 .. H*H/2-1
    const int v    = tid & 7;
    const int lane = (tid >> 3) & 31;
    const int subN = (tid >> 8) & 63;
    const int kt   = tid >> 14;

    const int o  = subN * 16 + (lane & 15);
    const int kp = kt * 16 + (v >> 2) * 8 + (lane >> 4) * 4 + (v & 3);
    const float scale = 1.0f / (float)NLAYER;

    float s0 = 0.f, s1 = 0.f;
    const float* p = conv_w + (size_t)o * HDIM + 2 * kp;
    #pragma unroll
    for (int l = 0; l < NLAYER; ++l) {
        float2 w = *(const float2*)(p + (size_t)l * HDIM * HDIM);
        s0 += w.x; s1 += w.y;
    }
    wfrag[tid] = pack_bf16(s0 * scale, s1 * scale);
}

// ---------------------------------------------------------------------------
// Kernel 2: fused GEMM (bf16 WMMA, f32 accum) + RMSNorm * norm_w.
// Block = 256 threads (8 wave32). Block owns 32 rows x all 1024 cols.
// Wave w owns cols [w*128, w*128+128) = 8 N-subtiles; 2 M-subtiles each.
// ---------------------------------------------------------------------------
__global__ void __launch_bounds__(256)
gemm_rmsnorm_kernel(const float* __restrict__ x,
                    const unsigned* __restrict__ wfrag,
                    const float* __restrict__ norm_w,
                    float* __restrict__ out) {
    __shared__ unsigned lds_x[32 * 16];   // 32 rows x 16 packed-kp dwords
    __shared__ float    lds_ss[32];       // per-row sum of squares

    const int tid  = threadIdx.x;
    const int wave = tid >> 5;
    const int lane = tid & 31;
    const int half = lane >> 4;           // 0: lanes 0-15, 1: lanes 16-31
    const int mrow = lane & 15;

    const int rowBase = blockIdx.x * 32;  // 32 rows of X / out
    const int waveN   = wave * 128;       // this wave's column base

    if (tid < 32) lds_ss[tid] = 0.f;

    const v8f vzero = {0.f, 0.f, 0.f, 0.f, 0.f, 0.f, 0.f, 0.f};
    v8f acc[2][8];
    #pragma unroll
    for (int m = 0; m < 2; ++m)
        #pragma unroll
        for (int s = 0; s < 8; ++s) acc[m][s] = vzero;

    // staging coords: thread t packs rows (t/16) and (t/16 + 16), kp = t%16
    const int sr = tid >> 4;
    const int sk = tid & 15;

    for (int kt = 0; kt < KTILES; ++kt) {
        __syncthreads();                  // protect lds_x reuse
        {
            const float* xp = x + (size_t)(rowBase + sr) * HDIM + kt * 32 + 2 * sk;
            float2 v0 = *(const float2*)xp;
            float2 v1 = *(const float2*)(xp + (size_t)16 * HDIM);
            lds_x[sr * 16 + sk]        = pack_bf16(v0.x, v0.y);
            lds_x[(sr + 16) * 16 + sk] = pack_bf16(v1.x, v1.y);
        }
        __syncthreads();

        // A fragments (16x32 bf16): lanes 0-15 row M=lane hold K 0..7 (v0-3)
        // and 16..23 (v4-7); lanes 16-31 hold K 8..15 and 24..31.
        v8u au0, au1;
        #pragma unroll
        for (int v = 0; v < 4; ++v) {
            au0[v]     = lds_x[mrow * 16 + half * 4 + v];
            au0[4 + v] = lds_x[mrow * 16 + 8 + half * 4 + v];
            au1[v]     = lds_x[(mrow + 16) * 16 + half * 4 + v];
            au1[4 + v] = lds_x[(mrow + 16) * 16 + 8 + half * 4 + v];
        }
        const v16bf a0 = __builtin_bit_cast(v16bf, au0);
        const v16bf a1 = __builtin_bit_cast(v16bf, au1);

        // B fragments: two 128-bit loads per subtile (fragment-major layout)
        const uint4* wf = (const uint4*)wfrag
                        + ((size_t)(kt * NSUB + wave * 8) * 32 + lane) * 2;
        #pragma unroll
        for (int sub = 0; sub < 8; ++sub) {
            uint4 lo = wf[sub * 64 + 0];
            uint4 hi = wf[sub * 64 + 1];
            v8u bu;
            bu[0] = lo.x; bu[1] = lo.y; bu[2] = lo.z; bu[3] = lo.w;
            bu[4] = hi.x; bu[5] = hi.y; bu[6] = hi.z; bu[7] = hi.w;
            const v16bf bfrag = __builtin_bit_cast(v16bf, bu);
            acc[0][sub] = __builtin_amdgcn_wmma_f32_16x16x32_bf16(
                false, a0, false, bfrag, (short)0, acc[0][sub], false, false);
            acc[1][sub] = __builtin_amdgcn_wmma_f32_16x16x32_bf16(
                false, a1, false, bfrag, (short)0, acc[1][sub], false, false);
        }
    }

    // ---- RMSNorm: reduce sum(y^2) per row across the whole block ----------
    // C-fragment mapping: element e -> row = m*16 + half*8 + e, col = mrow.
    float ss[2][8];
    #pragma unroll
    for (int m = 0; m < 2; ++m)
        #pragma unroll
        for (int e = 0; e < 8; ++e) ss[m][e] = 0.f;
    #pragma unroll
    for (int m = 0; m < 2; ++m)
        #pragma unroll
        for (int sub = 0; sub < 8; ++sub)
            #pragma unroll
            for (int e = 0; e < 8; ++e) {
                float v = acc[m][sub][e];
                ss[m][e] += v * v;
            }
    __syncthreads();
    #pragma unroll
    for (int m = 0; m < 2; ++m)
        #pragma unroll
        for (int e = 0; e < 8; ++e)
            atomicAdd(&lds_ss[m * 16 + half * 8 + e], ss[m][e]);
    __syncthreads();

    float rn[2][8];
    #pragma unroll
    for (int m = 0; m < 2; ++m)
        #pragma unroll
        for (int e = 0; e < 8; ++e)
            rn[m][e] = rsqrtf(lds_ss[m * 16 + half * 8 + e] * (1.0f / (float)HDIM) + EPS);

    #pragma unroll
    for (int sub = 0; sub < 8; ++sub) {
        const int col = waveN + sub * 16 + mrow;
        const float nw = norm_w[col];
        #pragma unroll
        for (int m = 0; m < 2; ++m)
            #pragma unroll
            for (int e = 0; e < 8; ++e) {
                const int grow = rowBase + m * 16 + half * 8 + e;
                out[(size_t)grow * HDIM + col] = acc[m][sub][e] * rn[m][e] * nw;
            }
    }
}

extern "C" void kernel_launch(void* const* d_in, const int* in_sizes, int n_in,
                              void* d_out, int out_size, void* d_ws, size_t ws_size,
                              hipStream_t stream) {
    const float* x      = (const float*)d_in[0];   // [B, S, H] f32
    const float* conv_w = (const float*)d_in[1];   // [L, H, H] f32
    const float* norm_w = (const float*)d_in[2];   // [H] f32
    float*       out    = (float*)d_out;           // [B, S, H] f32

    unsigned* wfrag = (unsigned*)d_ws;             // (H/2)*H dwords = 2 MB

    // 1) fold L layers + 1/L into fragment-major packed-bf16 B operand
    {
        const int total = (HDIM / 2) * HDIM;       // 524288
        reduce_w_kernel<<<total / 256, 256, 0, stream>>>(conv_w, wfrag);
    }
    // 2) fused bf16-WMMA GEMM + RMSNorm (32 rows x 1024 cols per block)
    {
        gemm_rmsnorm_kernel<<<ROWS / 32, 256, 0, stream>>>(x, wfrag, norm_w, out);
    }
}
